// CaptionNet_83184926589587
// MI455X (gfx1250) — matrix operand order repaired and checked by reading, not compiled, not run on hardware
//
#include <hip/hip_runtime.h>
#include <hip/hip_bf16.h>
#include <stdint.h>

// Problem dims
#define B_  256
#define T_  20
#define H_  512
#define WV_ 301
#define F_  196
#define C_  512
#define V_  9871

// Padded dims
#define KCAT  832    // 813 -> 26 k-tiles of 32
#define WVP   320    // 301 -> 10 k-tiles of 32 (lstm input K)
#define FP64  256    // attn N (196) padded to 64-mult
#define WVP64 320    // comb N (301) padded to 64-mult
#define VP64  9920   // vocab N (9871) padded to 64-mult

typedef __bf16 bfloat;
typedef __attribute__((ext_vector_type(16))) __bf16 v16bf;
typedef __attribute__((ext_vector_type(8)))  __bf16 v8bf;
typedef __attribute__((ext_vector_type(8)))  float  v8f;

// ---------------------------------------------------------------------------
// A-fragment: 16x32 bf16, M = lane&15, K contiguous (matches B):
//   lanes 0-15:  j=0..15 -> K=j
//   lanes 16-31: j=0..15 -> K=16+j
// => two contiguous 8-element (16B) chunks at k0+16*hi and k0+16*hi+8.
__device__ __forceinline__ v16bf load_a_frag(const bfloat* __restrict__ A, int lda,
                                             int m0, int k0, int lane) {
  int row = m0 + (lane & 15);
  int hi  = lane >> 4;
  const bfloat* p = A + (long)row * lda + k0 + 16 * hi;
  v8bf lo = *(const v8bf*)(p);
  v8bf up = *(const v8bf*)(p + 8);
  v16bf a;
#pragma unroll
  for (int i = 0; i < 8; ++i) { a[i] = lo[i]; a[i + 8] = up[i]; }
  return a;
}

// B-fragment: 32x16 bf16 of W^T, i.e. B[k][n] = W[n0+n][k0+k].
// N = lane&15, lanes 0-15 hold K=0..15, lanes 16-31 hold K=16..31
// => one contiguous 16-element (32B) read of row n at k0+16*hi.
__device__ __forceinline__ v16bf load_b_frag(const bfloat* __restrict__ W, int ldw,
                                             int n0, int k0, int lane) {
  int row = n0 + (lane & 15);
  int hi  = lane >> 4;
  return *(const v16bf*)(W + (long)row * ldw + k0 + 16 * hi);
}

// ---------------------------------------------------------------------------
// Generic GEMM: out = act( A1 @ W1^T (+ A2 @ W2^T) + bias1 (+ bias2) )
// A: [256, lda] bf16 row-major; W: [Npad64, ldw] bf16 row-major (zero-padded).
// Wave computes a 16x64 strip. 8 waves / 256-thread block.
// Software-pipelined: tile k+1 fragments are loaded before tile k's WMMAs so
// the load latency overlaps the matrix math (partial s_wait_loadcnt).
// grid.x = 2 * nTiles64  (16 m-tiles * nTiles64 n-strips / 8 waves per block)
__global__ __launch_bounds__(256) void wmma_gemm_kernel(
    const bfloat* __restrict__ A1, int lda1,
    const bfloat* __restrict__ W1, int ldw1, int k1Tiles,
    const bfloat* __restrict__ A2, int lda2,
    const bfloat* __restrict__ W2, int ldw2, int k2Tiles,
    const float* __restrict__ bias1, const float* __restrict__ bias2,
    float* __restrict__ outF, bfloat* __restrict__ outB, long ldo,
    int nTiles64, int Nreal, int act)  // act: 0=none 1=relu 2=sigmoid
{
  int wave = (int)((blockIdx.x * blockDim.x + threadIdx.x) >> 5);
  int lane = (int)(threadIdx.x & 31);
  int mt   = wave / nTiles64;
  int n64  = wave % nTiles64;
  int m0   = mt * 16;
  int n0   = n64 * 64;

  v8f acc[4] = {};

  auto mm = [&](const bfloat* __restrict__ A, int lda,
                const bfloat* __restrict__ W, int ldw, int kTiles) {
    // prologue: fragments for tile 0
    v16bf a0 = load_a_frag(A, lda, m0, 0, lane);
    v16bf b0[4];
#pragma unroll
    for (int j = 0; j < 4; ++j) b0[j] = load_b_frag(W, ldw, n0 + j * 16, 0, lane);

#pragma unroll 2
    for (int kt = 0; kt < kTiles; ++kt) {
      // prefetch tile kt+1 (clamped => always in-bounds, branch-free)
      int kn = (kt + 1 < kTiles) ? (kt + 1) * 32 : kt * 32;
      v16bf a1 = load_a_frag(A, lda, m0, kn, lane);
      v16bf b1[4];
#pragma unroll
      for (int j = 0; j < 4; ++j) b1[j] = load_b_frag(W, ldw, n0 + j * 16, kn, lane);

      // compute tile kt (its loads were issued last iteration)
#pragma unroll
      for (int j = 0; j < 4; ++j)
        acc[j] = __builtin_amdgcn_wmma_f32_16x16x32_bf16(
            false, a0, false, b0[j], (short)0, acc[j], false, false);

      a0 = a1;
#pragma unroll
      for (int j = 0; j < 4; ++j) b0[j] = b1[j];
    }
  };

  mm(A1, lda1, W1, ldw1, k1Tiles);
  if (A2) mm(A2, lda2, W2, ldw2, k2Tiles);

  // C/D layout: VGPR v -> M = v + 8*(lane>=16), N = lane&15
  int hi = lane >> 4;
  int nc = lane & 15;
#pragma unroll
  for (int j = 0; j < 4; ++j) {
    int n = n0 + j * 16 + nc;
    float bv = 0.f;
    if (n < Nreal) {
      bv = bias1[n];
      if (bias2) bv += bias2[n];
    }
#pragma unroll
    for (int v = 0; v < 8; ++v) {
      int m = m0 + v + 8 * hi;
      float val = acc[j][v] + bv;
      if (act == 1) val = fmaxf(val, 0.f);
      else if (act == 2) val = 1.f / (1.f + __expf(-val));
      if (n < Nreal) {
        if (outF) outF[(long)m * ldo + n] = val;
        else      outB[(long)m * ldo + n] = (bfloat)val;
      } else if (outB && n < (int)ldo) {
        outB[(long)m * ldo + n] = (bfloat)0.f;  // keep bf16 pad columns zero
      }
    }
  }
}

// ---------------------------------------------------------------------------
// f32 -> bf16 weight convert with zero padding. dst is [Npad, Kpad].
__global__ void cvt_pad_kernel(const float* __restrict__ src, bfloat* __restrict__ dst,
                               int Nreal, int Kreal, int Npad, int Kpad) {
  long id = (long)blockIdx.x * blockDim.x + threadIdx.x;
  long total = (long)Npad * Kpad;
  if (id >= total) return;
  int n = (int)(id / Kpad);
  int k = (int)(id % Kpad);
  float v = (n < Nreal && k < Kreal) ? src[(long)n * Kreal + k] : 0.f;
  dst[id] = (bfloat)v;
}

// encoding [B,C,F] f32 -> enc_t [B,F,C] bf16 (coalesced for ctx matvec)
__global__ void enc_cvt_kernel(const float* __restrict__ enc, bfloat* __restrict__ enc_t) {
  long id = (long)blockIdx.x * blockDim.x + threadIdx.x;
  long total = (long)B_ * F_ * C_;
  if (id >= total) return;
  int c = (int)(id % C_);
  long r = id / C_;
  int f = (int)(r % F_);
  int b = (int)(r / F_);
  enc_t[id] = (bfloat)enc[((long)b * C_ + c) * F_ + f];
}

__global__ void zero_u32_kernel(uint32_t* __restrict__ p, long n) {
  long i = (long)blockIdx.x * blockDim.x + threadIdx.x;
  if (i < n) p[i] = 0u;
}

// x_t = (t==0 ? 0 : wordvecs[:,t-1,:]); write bf16 x into xh[:,0:301] and
// xc[:,0:301]; write previous h (bf16) into xh[:,301:813].
__global__ void build_xh_kernel(const float* __restrict__ wordvecs,
                                const bfloat* __restrict__ hbf,
                                bfloat* __restrict__ xh, bfloat* __restrict__ xc, int t) {
  int id = (int)(blockIdx.x * blockDim.x + threadIdx.x);
  if (id >= B_ * 813) return;
  int b = id / 813;
  int k = id % 813;
  if (k < WV_) {
    float xv = (t == 0) ? 0.f : wordvecs[((long)b * T_ + (t - 1)) * WV_ + k];
    bfloat bv = (bfloat)xv;
    xh[(long)b * KCAT + k] = bv;
    xc[(long)b * KCAT + k] = bv;
  } else {
    xh[(long)b * KCAT + k] = hbf[(long)b * H_ + (k - WV_)];
  }
}

// In-place row softmax over 196 elements. One 256-thread block per row.
__global__ void softmax196_kernel(float* __restrict__ aw) {
  __shared__ float red[256];
  int b = blockIdx.x, t = threadIdx.x;
  float v = (t < F_) ? aw[(long)b * F_ + t] : -3.0e38f;
  red[t] = v; __syncthreads();
  for (int s = 128; s > 0; s >>= 1) { if (t < s) red[t] = fmaxf(red[t], red[t + s]); __syncthreads(); }
  float mx = red[0]; __syncthreads();
  float e = (t < F_) ? __expf(v - mx) : 0.f;
  red[t] = e; __syncthreads();
  for (int s = 128; s > 0; s >>= 1) { if (t < s) red[t] += red[t + s]; __syncthreads(); }
  float inv = 1.f / red[0];
  if (t < F_) aw[(long)b * F_ + t] = e * inv;
}

// ctx[b,c] = gamma[b,c] * sum_f aw[b,f] * enc_t[b,f,c]; write bf16 into xc[:,301:813].
__global__ void ctx_kernel(const bfloat* __restrict__ enc_t, const float* __restrict__ aw,
                           const float* __restrict__ gamma, bfloat* __restrict__ xc) {
  __shared__ float s_aw[F_];
  int b = blockIdx.x;
  int c = threadIdx.x;  // 512 threads
  for (int f = threadIdx.x; f < F_; f += blockDim.x) s_aw[f] = aw[(long)b * F_ + f];
  __syncthreads();
  const bfloat* e = enc_t + (long)b * F_ * C_ + c;
  float acc = 0.f;
#pragma unroll 4
  for (int f = 0; f < F_; ++f) acc += s_aw[f] * (float)e[(long)f * C_];
  float g = gamma[(long)b * C_ + c];
  xc[(long)b * KCAT + WV_ + c] = (bfloat)(g * acc);
}

// LSTM pointwise cell: gates [B,4H] (i,f,g,o); c stays f32, h kept as bf16.
__global__ void lstm_cell_kernel(const float* __restrict__ gates,
                                 float* __restrict__ c, bfloat* __restrict__ hbf) {
  int b = blockIdx.x;
  int j = threadIdx.x;  // 512
  const float* g = gates + (long)b * 4 * H_;
  float gi = g[j], gf = g[H_ + j], gg = g[2 * H_ + j], go = g[3 * H_ + j];
  float si = 1.f / (1.f + __expf(-gi));
  float sf = 1.f / (1.f + __expf(-gf));
  float so = 1.f / (1.f + __expf(-go));
  float cn = sf * c[(long)b * H_ + j] + si * tanhf(gg);
  c[(long)b * H_ + j] = cn;
  hbf[(long)b * H_ + j] = (bfloat)(so * tanhf(cn));
}

// ---------------------------------------------------------------------------
static inline void launch_gemm(hipStream_t s,
                               const bfloat* A1, int lda1, const bfloat* W1, int ldw1, int k1,
                               const bfloat* A2, int lda2, const bfloat* W2, int ldw2, int k2,
                               const float* b1, const float* b2,
                               float* oF, bfloat* oB, long ldo,
                               int nT64, int Nreal, int act) {
  wmma_gemm_kernel<<<dim3(2 * nT64), dim3(256), 0, s>>>(
      A1, lda1, W1, ldw1, k1, A2, lda2, W2, ldw2, k2, b1, b2, oF, oB, ldo, nT64, Nreal, act);
}

extern "C" void kernel_launch(void* const* d_in, const int* in_sizes, int n_in,
                              void* d_out, int out_size, void* d_ws, size_t ws_size,
                              hipStream_t stream) {
  (void)in_sizes; (void)n_in; (void)out_size; (void)ws_size;
  const float* encoding = (const float*)d_in[0];
  const float* wordvecs = (const float*)d_in[1];
  const float* attn_W   = (const float*)d_in[2];
  const float* attn_b   = (const float*)d_in[3];
  const float* comb_W   = (const float*)d_in[4];
  const float* comb_b   = (const float*)d_in[5];
  const float* gate_W   = (const float*)d_in[6];
  const float* gate_b   = (const float*)d_in[7];
  const float* lstm_Wih = (const float*)d_in[8];
  const float* lstm_Whh = (const float*)d_in[9];
  const float* lstm_bih = (const float*)d_in[10];
  const float* lstm_bhh = (const float*)d_in[11];
  const float* vocab_W  = (const float*)d_in[12];
  const float* vocab_b  = (const float*)d_in[13];
  float* out = (float*)d_out;

  // ---- carve workspace (all 256B aligned) ----
  char* ws = (char*)d_ws;
  size_t off = 0;
  auto carve = [&](size_t bytes) -> char* {
    char* p = ws + off;
    off += (bytes + 255) & ~(size_t)255;
    return p;
  };
  bfloat* enc_t    = (bfloat*)carve((size_t)B_ * F_ * C_ * 2);
  bfloat* attn_Wb  = (bfloat*)carve((size_t)FP64 * KCAT * 2);
  bfloat* comb_Wb  = (bfloat*)carve((size_t)WVP64 * KCAT * 2);
  bfloat* gate_Wb  = (bfloat*)carve((size_t)H_ * H_ * 2);
  bfloat* wih_b    = (bfloat*)carve((size_t)(4 * H_) * WVP * 2);
  bfloat* whh_b    = (bfloat*)carve((size_t)(4 * H_) * H_ * 2);
  bfloat* vocab_Wb = (bfloat*)carve((size_t)VP64 * H_ * 2);
  char*   state0   = ws + off;                               // zeroed region start
  bfloat* h_bf     = (bfloat*)carve((size_t)B_ * H_ * 2);
  float*  c_st     = (float*)carve((size_t)B_ * H_ * 4);
  bfloat* xh       = (bfloat*)carve((size_t)B_ * KCAT * 2);
  bfloat* xc       = (bfloat*)carve((size_t)B_ * KCAT * 2);
  bfloat* inp_bf   = (bfloat*)carve((size_t)B_ * WVP * 2);
  char*   state1   = ws + off;                               // zeroed region end
  float*  aw       = (float*)carve((size_t)B_ * F_ * 4);
  float*  gamma    = (float*)carve((size_t)B_ * C_ * 4);
  float*  gates    = (float*)carve((size_t)B_ * 4 * H_ * 4);

  // ---- one-time (per call) init: zero state, convert weights/encoding ----
  {
    long nz = (long)(state1 - state0) / 4;
    zero_u32_kernel<<<dim3((unsigned)((nz + 255) / 256)), dim3(256), 0, stream>>>(
        (uint32_t*)state0, nz);
  }
  auto cvt = [&](const float* src, bfloat* dst, int Nr, int Kr, int Np, int Kp) {
    long total = (long)Np * Kp;
    cvt_pad_kernel<<<dim3((unsigned)((total + 255) / 256)), dim3(256), 0, stream>>>(
        src, dst, Nr, Kr, Np, Kp);
  };
  cvt(attn_W,   attn_Wb,  F_,     813, FP64,   KCAT);
  cvt(comb_W,   comb_Wb,  WV_,    813, WVP64,  KCAT);
  cvt(gate_W,   gate_Wb,  C_,     H_,  H_,     H_);
  cvt(lstm_Wih, wih_b,    4 * H_, WV_, 4 * H_, WVP);
  cvt(lstm_Whh, whh_b,    4 * H_, H_,  4 * H_, H_);
  cvt(vocab_W,  vocab_Wb, V_,     H_,  VP64,   H_);
  {
    long total = (long)B_ * F_ * C_;
    enc_cvt_kernel<<<dim3((unsigned)((total + 255) / 256)), dim3(256), 0, stream>>>(
        encoding, enc_t);
  }

  // ---- time loop (sequential dependence) ----
  for (int t = 0; t < T_; ++t) {
    // 1) assemble [x, h_prev] (bf16) and x part of [x, ctx]
    build_xh_kernel<<<dim3((B_ * 813 + 255) / 256), dim3(256), 0, stream>>>(
        wordvecs, h_bf, xh, xc, t);
    // 2) attention logits: aw = xh @ attn_W^T + attn_b   [256,196]
    launch_gemm(stream, xh, KCAT, attn_Wb, KCAT, KCAT / 32,
                nullptr, 0, nullptr, 0, 0, attn_b, nullptr,
                aw, nullptr, F_, FP64 / 64, F_, 0);
    // 3) softmax over 196
    softmax196_kernel<<<dim3(B_), dim3(256), 0, stream>>>(aw);
    // 4) gate: gamma = sigmoid(h @ gate_W^T + gate_b)    [256,512]
    launch_gemm(stream, h_bf, H_, gate_Wb, H_, H_ / 32,
                nullptr, 0, nullptr, 0, 0, gate_b, nullptr,
                gamma, nullptr, C_, H_ / 64, C_, 2);
    // 5) gated context -> xc[:,301:813]
    ctx_kernel<<<dim3(B_), dim3(512), 0, stream>>>(enc_t, aw, gamma, xc);
    // 6) comb: inp = relu(xc @ comb_W^T + comb_b) -> bf16 [256,320]
    launch_gemm(stream, xc, KCAT, comb_Wb, KCAT, KCAT / 32,
                nullptr, 0, nullptr, 0, 0, comb_b, nullptr,
                nullptr, inp_bf, WVP, WVP64 / 64, WV_, 1);
    // 7) gates = inp @ Wih^T + h @ Whh^T + bih + bhh     [256,2048]
    launch_gemm(stream, inp_bf, WVP, wih_b, WVP, WVP / 32,
                h_bf, H_, whh_b, H_, H_ / 32, lstm_bih, lstm_bhh,
                gates, nullptr, 4 * H_, (4 * H_) / 64, 4 * H_, 0);
    // 8) LSTM cell pointwise -> c (f32), h (bf16)
    lstm_cell_kernel<<<dim3(B_), dim3(512), 0, stream>>>(gates, c_st, h_bf);
    // 9) logits = h @ vocab_W^T + vocab_b -> d_out[:, t, :]
    launch_gemm(stream, h_bf, H_, vocab_Wb, H_, H_ / 32,
                nullptr, 0, nullptr, 0, 0, vocab_b, nullptr,
                out + (long)t * V_, nullptr, (long)T_ * V_, VP64 / 64, V_, 0);
  }
}
